// GeometricAttention_21139829031332
// MI455X (gfx1250) — compile-verified, matrix operand order checked by reading, not run on hardware
//
#include <hip/hip_runtime.h>
#include <hip/hip_bf16.h>

typedef __attribute__((ext_vector_type(16))) _Float16 v16h;
typedef __attribute__((ext_vector_type(8)))  float    v8f;

#define EMBED 256
#define NHEAD 8
#define HDIM  32
#define BATCH 4
#define SEQ   2048

union F16Frag { v16h v; _Float16 h[16]; uint4 q[2]; };

// A-fragment (16x32 f16, MxK) or B-fragment (32x16 f16, KxN loaded from a
// K-contiguous "Bt" layout). Per ISA 7.12.2: lanes 0-15 hold M(or N)=lane,
// VGPR0..3 = K 0..7 packed, VGPR4..7 = K 16..23; lanes 16-31 shift K by 8.
__device__ inline v16h load_frag_f16(const _Float16* base, int ld) {
  int lane = threadIdx.x & 31;
  int m = lane & 15, hh = lane >> 4;
  const _Float16* p = base + m * ld + hh * 8;
  F16Frag f;
  f.q[0] = *(const uint4*)(p);        // K = hh*8 .. hh*8+7
  f.q[1] = *(const uint4*)(p + 16);   // K = 16+hh*8 .. 16+hh*8+7
  return f.v;
}

__device__ inline v8f wmma_f16(v16h a, v16h b, v8f c) {
  return __builtin_amdgcn_wmma_f32_16x16x32_f16(false, a, false, b,
                                                (short)0, c, false, false);
}

// ---------------------------------------------------------------------------
// Kernel 0: convert x -> f16, and transpose/convert weights to f16 "Bt"
// layouts (N x K, K contiguous).
// ---------------------------------------------------------------------------
__global__ void prep_inputs(const float* __restrict__ x,
                            const float* __restrict__ qkv_w,
                            const float* __restrict__ proj_w,
                            _Float16* __restrict__ xh,
                            _Float16* __restrict__ qkv_wt,
                            _Float16* __restrict__ proj_wt) {
  int i = blockIdx.x * blockDim.x + threadIdx.x;
  if (i < BATCH * SEQ * EMBED) xh[i] = (_Float16)x[i];
  if (i < 768 * 256) {
    int n = i >> 8, k = i & 255;                 // qkv_w is (256, 768)
    qkv_wt[n * 256 + k] = (_Float16)qkv_w[k * 768 + n];
  }
  if (i < 256 * 256) {
    int n = i >> 8, k = i & 255;                 // proj_w is (256, 256)
    proj_wt[n * 256 + k] = (_Float16)proj_w[k * 256 + n];
  }
}

// ---------------------------------------------------------------------------
// Kernel 1: QKV projection.  (B*N, 256) x (256, 768) + bias.
// Q,K -> [B][H][N][D] f16 ; V -> [B][H][D][N] f16 (pre-transposed for AV).
// One wave = one 16x16 output tile, K=256 in 8 WMMA steps.
// ---------------------------------------------------------------------------
__global__ __launch_bounds__(256) void qkv_gemm(
    const _Float16* __restrict__ xh, const _Float16* __restrict__ wt,
    const float* __restrict__ qkv_b, _Float16* __restrict__ qbuf,
    _Float16* __restrict__ kbuf, _Float16* __restrict__ vbuf) {
  int wv = threadIdx.x >> 5;
  int lane = threadIdx.x & 31;
  int m0 = (blockIdx.x * 8 + wv) * 16;   // token-row tile
  int n0 = blockIdx.y * 16;              // output-column tile (0..767)
  v8f acc = {};
#pragma unroll
  for (int k0 = 0; k0 < 256; k0 += 32) {
    v16h a = load_frag_f16(xh + (size_t)m0 * 256 + k0, 256);
    v16h b = load_frag_f16(wt + (size_t)n0 * 256 + k0, 256);
    acc = wmma_f16(a, b, acc);
  }
  int n = lane & 15, hh = lane >> 4;
  int c = n0 + n;
  int which = c >> 8;          // 0=q 1=k 2=v
  int cc = c & 255;
  int head = cc >> 5, d = cc & 31;
  float bias = qkv_b[c];
#pragma unroll
  for (int j = 0; j < 8; ++j) {
    int row = m0 + j + 8 * hh;           // global token row in [0, 8192)
    int b_ = row >> 11, tok = row & 2047;
    _Float16 hv = (_Float16)(acc[j] + bias);
    if (which == 0)
      qbuf[(((size_t)(b_ * 8 + head)) * SEQ + tok) * HDIM + d] = hv;
    else if (which == 1)
      kbuf[(((size_t)(b_ * 8 + head)) * SEQ + tok) * HDIM + d] = hv;
    else
      vbuf[(((size_t)(b_ * 8 + head)) * HDIM + d) * SEQ + tok] = hv;
  }
}

// ---------------------------------------------------------------------------
// Kernel 2: flash attention with distance bias.
// A block's 8 waves share (b,h); keys stream in 32-key chunks that are
// async-copied (GLOBAL_LOAD_ASYNC_TO_LDS_B128, double-buffered) into LDS
// once per block. Per chunk: 2 WMMAs for scores (K-dim = D = 32), online
// softmax with width-16 shuffle reductions, probs staged via LDS into an
// A-fragment, 2 WMMAs for AV.
// ---------------------------------------------------------------------------
__global__ __launch_bounds__(256) void attn_kernel(
    const _Float16* __restrict__ qbuf, const _Float16* __restrict__ kbuf,
    const _Float16* __restrict__ vbuf, const float* __restrict__ distances,
    const float* __restrict__ dist_w, const float* __restrict__ dist_b,
    _Float16* __restrict__ obuf) {
  // [buf][0..1023] = K tile (32 keys x 32 d), [buf][1024..2047] = V tile
  // (32 d-rows x 32 keys). Double buffered => 8 KB.
  __shared__ _Float16 kv[2][2048];
  __shared__ _Float16 pstage[8][512];    // per-wave 16x32 prob staging (8 KB)

  int tid = threadIdx.x;
  int wv = tid >> 5, lane = tid & 31;
  int gid = blockIdx.x * 8 + wv;         // 4096 waves total
  int qt = gid & 127;                    // query tile 0..127 (same b,h per block)
  int h  = (gid >> 7) & 7;
  int b  = gid >> 10;
  int n = lane & 15, hh = lane >> 4;
  const float inv_sqrt_d = 0.17677669529663687f;  // 1/sqrt(32)
  float dw = dist_w[h], db = dist_b[h];

  const _Float16* qptr = qbuf + ((size_t)(b * 8 + h) * SEQ + qt * 16) * HDIM;
  const _Float16* kptr = kbuf + (size_t)(b * 8 + h) * SEQ * HDIM;
  const _Float16* vptr = vbuf + (size_t)(b * 8 + h) * HDIM * SEQ;
  const float* dptr = distances + (size_t)b * SEQ * SEQ + (size_t)(qt * 16) * SEQ;

  // Async-copy one 32-key chunk (K: 2 KB contiguous; V: 32 rows x 64 B)
  // into kv[sel]. 256 threads x 16 B = 4 KB.
  auto issue_copy = [&](int kc, int sel) {
    const char* src;
    unsigned ldsoff;
    if (tid < 128) {
      src = (const char*)(kptr + (size_t)kc * HDIM) + tid * 16;
      ldsoff = (unsigned)(uintptr_t)(&kv[sel][0]) + tid * 16;
    } else {
      int i = tid - 128;
      int r = i >> 2, seg = i & 3;       // d-row, 16B segment
      src = (const char*)(vptr + (size_t)r * SEQ + kc) + seg * 16;
      ldsoff = (unsigned)(uintptr_t)(&kv[sel][1024]) + r * 64 + seg * 16;
    }
    asm volatile("global_load_async_to_lds_b128 %0, %1, off"
                 :: "v"(ldsoff), "v"((unsigned long long)(uintptr_t)src)
                 : "memory");
  };

  v16h qf = load_frag_f16(qptr, HDIM);
  v8f o0 = {}, o1 = {};
  float rmax[8], rsum[8];
#pragma unroll
  for (int j = 0; j < 8; ++j) { rmax[j] = -3.0e38f; rsum[j] = 0.f; }

  issue_copy(0, 0);
  for (int kc = 0; kc < SEQ; kc += 32) {
    int sel = (kc >> 5) & 1;
    asm volatile("s_wait_asynccnt 0" ::: "memory");
    __syncthreads();                     // chunk kc landed; buf sel^1 free
    if (kc + 32 < SEQ) issue_copy(kc + 32, sel ^ 1);

    const _Float16* kt = &kv[sel][0];
    const _Float16* vt = &kv[sel][1024];
    v16h kb0 = load_frag_f16(kt, 32);            // keys kc..kc+15
    v16h kb1 = load_frag_f16(kt + 16 * 32, 32);  // keys kc+16..kc+31
    v8f zero = {};
    v8f s0 = wmma_f16(qf, kb0, zero);
    v8f s1 = wmma_f16(qf, kb1, zero);

#pragma unroll
    for (int j = 0; j < 8; ++j) {
      int row = j + 8 * hh;              // row within query tile
      const float* dp = dptr + (size_t)row * SEQ + kc + n;
      __builtin_prefetch(dp + 32, 0, 0);            // stream next dist lines
      float sc0 = fmaf(s0[j], inv_sqrt_d, fmaf(dp[0], dw, db));
      float sc1 = fmaf(s1[j], inv_sqrt_d, fmaf(dp[16], dw, db));
      float tmax = fmaxf(sc0, sc1);
#pragma unroll
      for (int msk = 1; msk < 16; msk <<= 1)
        tmax = fmaxf(tmax, __shfl_xor(tmax, msk, 16));
      float mnew = fmaxf(rmax[j], tmax);
      float a = __expf(rmax[j] - mnew);
      rmax[j] = mnew;
      float p0 = __expf(sc0 - mnew);
      float p1 = __expf(sc1 - mnew);
      float ps = p0 + p1;
#pragma unroll
      for (int msk = 1; msk < 16; msk <<= 1)
        ps += __shfl_xor(ps, msk, 16);
      rsum[j] = rsum[j] * a + ps;
      o0[j] *= a;
      o1[j] *= a;
      pstage[wv][row * 32 + n]      = (_Float16)p0;
      pstage[wv][row * 32 + 16 + n] = (_Float16)p1;
    }
    asm volatile("s_wait_dscnt 0" ::: "memory");    // wave-local LDS RAW
    v16h pf  = load_frag_f16(&pstage[wv][0], 32);   // probs as A-fragment
    v16h vb0 = load_frag_f16(vt, 32);               // d = 0..15
    v16h vb1 = load_frag_f16(vt + 16 * 32, 32);     // d = 16..31
    o0 = wmma_f16(pf, vb0, o0);
    o1 = wmma_f16(pf, vb1, o1);
  }

#pragma unroll
  for (int j = 0; j < 8; ++j) {
    int row = qt * 16 + j + 8 * hh;
    float inv = 1.0f / rsum[j];
    size_t base = ((size_t)(b * SEQ + row)) * EMBED + h * HDIM;
    obuf[base + n]      = (_Float16)(o0[j] * inv);
    obuf[base + 16 + n] = (_Float16)(o1[j] * inv);
  }
}

// ---------------------------------------------------------------------------
// Kernel 3: output projection (8192x256) x (256x256) + bias -> fp32 d_out
// ---------------------------------------------------------------------------
__global__ __launch_bounds__(256) void proj_gemm(
    const _Float16* __restrict__ obuf, const _Float16* __restrict__ pwt,
    const float* __restrict__ proj_b, float* __restrict__ out) {
  int wv = threadIdx.x >> 5, lane = threadIdx.x & 31;
  int m0 = (blockIdx.x * 8 + wv) * 16;
  int n0 = blockIdx.y * 16;
  v8f acc = {};
#pragma unroll
  for (int k0 = 0; k0 < 256; k0 += 32) {
    v16h a = load_frag_f16(obuf + (size_t)m0 * 256 + k0, 256);
    v16h b = load_frag_f16(pwt + (size_t)n0 * 256 + k0, 256);
    acc = wmma_f16(a, b, acc);
  }
  int n = lane & 15, hh = lane >> 4;
  float bias = proj_b[n0 + n];
#pragma unroll
  for (int j = 0; j < 8; ++j) {
    int row = m0 + j + 8 * hh;
    out[(size_t)row * 256 + n0 + n] = acc[j] + bias;
  }
}

extern "C" void kernel_launch(void* const* d_in, const int* in_sizes, int n_in,
                              void* d_out, int out_size, void* d_ws, size_t ws_size,
                              hipStream_t stream) {
  const float* x         = (const float*)d_in[0];
  const float* distances = (const float*)d_in[1];
  const float* qkv_w     = (const float*)d_in[2];
  const float* qkv_b     = (const float*)d_in[3];
  const float* proj_w    = (const float*)d_in[4];
  const float* proj_b    = (const float*)d_in[5];
  const float* dist_w    = (const float*)d_in[6];
  const float* dist_b    = (const float*)d_in[7];

  // Workspace carve-up (f16 elements).
  const size_t qkvsz = (size_t)BATCH * NHEAD * SEQ * HDIM;   // 2,097,152
  _Float16* ws      = (_Float16*)d_ws;
  _Float16* qbuf    = ws;
  _Float16* kbuf    = qbuf + qkvsz;
  _Float16* vbuf    = kbuf + qkvsz;            // stored [B][H][D][N]
  _Float16* obuf    = vbuf + qkvsz;            // stored [B][N][H*D]
  _Float16* xh      = obuf + qkvsz;            // x converted to f16
  _Float16* qkv_wt  = xh + qkvsz;              // 768 x 256
  _Float16* proj_wt = qkv_wt + 768 * 256;      // 256 x 256

  // covers x conversion (2,097,152) and both weight transposes
  prep_inputs<<<dim3((BATCH * SEQ * EMBED + 255) / 256), dim3(256), 0, stream>>>(
      x, qkv_w, proj_w, xh, qkv_wt, proj_wt);

  // M tiles = 8192/16 = 512 (8 per block), N tiles = 768/16 = 48
  qkv_gemm<<<dim3(64, 48), dim3(256), 0, stream>>>(xh, qkv_wt, qkv_b,
                                                   qbuf, kbuf, vbuf);

  // 4096 waves (b,h,qtile), 8 waves per block, block-shared (b,h)
  attn_kernel<<<dim3(512), dim3(256), 0, stream>>>(qbuf, kbuf, vbuf, distances,
                                                   dist_w, dist_b, obuf);

  // M tiles = 512, N tiles = 256/16 = 16
  proj_gemm<<<dim3(64, 16), dim3(256), 0, stream>>>(obuf, proj_wt, proj_b,
                                                    (float*)d_out);
}
